// MultibandFrameAttention_18210661335567
// MI455X (gfx1250) — compile-verified, matrix operand order checked by reading, not run on hardware
//
#include <hip/hip_runtime.h>

typedef __attribute__((ext_vector_type(16))) __bf16 v16bf;
typedef __attribute__((ext_vector_type(8)))  float  v8f;

#define BINS   512
#define CROP   1024
#define NBANDS 8
#define HD     64
#define NBATCH 4
#define ROWS   (NBATCH * CROP)   // 4096

static __device__ __forceinline__ __bf16 bfc(float x) { return (__bf16)x; }

// K index of packed pair p (0..7) for the 16-bit A/B WMMA fragment layout:
// VGPR v in 0..3 -> K base 0 (+half*8), VGPR 4..7 -> K base 16 (+half*8); pairs contiguous.
static __device__ __forceinline__ int kidx(int p, int half) {
    return ((p >> 2) << 4) + (half << 3) + ((p & 3) << 1);
}

static __device__ __forceinline__ v8f wmma_bf16(v16bf a, v16bf b, v8f c) {
    return __builtin_amdgcn_wmma_f32_16x16x32_bf16(false, a, false, b, (short)0, c,
                                                   false, false);
}

// ---------------------------------------------------------------------------
// Y[4096,512] = X[4096,512] * W[512,512]^T + bias   (bf16 WMMA, fp32 accum)
// One wave computes a 16(M) x 64(N) strip; k-chunks software-pipelined
// (double-buffered bf16 fragments) so loads of chunk k+1 overlap the 4 WMMAs
// of chunk k. blockDim = 128 (4 waves).
// ---------------------------------------------------------------------------
__global__ __launch_bounds__(128)
void mba_gemm_bias_wmma(const float* __restrict__ X, const float* __restrict__ W,
                        const float* __restrict__ bias, float* __restrict__ Y)
{
    const int lane = threadIdx.x & 31;
    const int wid  = threadIdx.x >> 5;
    const int half = lane >> 4;
    const int l16  = lane & 15;

    const int job = blockIdx.x * 4 + wid;   // 2048 jobs
    const int m0  = (job & 255) << 4;       // 256 m-tiles
    const int n0  = (job >> 8) << 6;        // 8 n-strips

    const float* xrow = X + (size_t)(m0 + l16) * BINS;   // A: lane holds row m0+l16
    const float* wrow = W + (size_t)(n0 + l16) * BINS;   // B: lane holds col n0+t*16+l16

    auto loadA = [&](int k0, v16bf& a) {
#pragma unroll
        for (int p = 0; p < 8; ++p) {
            const int k = k0 + kidx(p, half);
            const float2 f = *(const float2*)(xrow + k);
            a[2 * p] = bfc(f.x); a[2 * p + 1] = bfc(f.y);
        }
    };
    auto loadB = [&](int k0, v16bf* b) {
#pragma unroll
        for (int t = 0; t < 4; ++t) {
            const float* wr = wrow + (size_t)(t * 16) * BINS;
#pragma unroll
            for (int p = 0; p < 8; ++p) {
                const int k = k0 + kidx(p, half);
                const float2 f = *(const float2*)(wr + k);
                b[t][2 * p] = bfc(f.x); b[t][2 * p + 1] = bfc(f.y);
            }
        }
    };

    v8f acc[4] = {{}, {}, {}, {}};
    v16bf a_cur, b_cur[4];
    loadA(0, a_cur);
    loadB(0, b_cur);

    for (int k0 = 0; k0 < BINS - 32; k0 += 32) {
        // prefetch one chunk beyond the one we are about to load (L2 hint)
        if (k0 + 64 < BINS) {
            __builtin_prefetch(xrow + k0 + 64, 0, 0);
            __builtin_prefetch(wrow + (size_t)(half * 32) * BINS + k0 + 64, 0, 0);
        }
        v16bf a_nxt, b_nxt[4];
        loadA(k0 + 32, a_nxt);
        loadB(k0 + 32, b_nxt);
#pragma unroll
        for (int t = 0; t < 4; ++t)
            acc[t] = wmma_bf16(a_cur, b_cur[t], acc[t]);
        a_cur = a_nxt;
#pragma unroll
        for (int t = 0; t < 4; ++t) b_cur[t] = b_nxt[t];
    }
#pragma unroll
    for (int t = 0; t < 4; ++t)
        acc[t] = wmma_bf16(a_cur, b_cur[t], acc[t]);

#pragma unroll
    for (int t = 0; t < 4; ++t) {
        const int n = n0 + t * 16 + l16;
        const float bv = bias[n];
#pragma unroll
        for (int r = 0; r < 8; ++r) {
            const int m = m0 + r + 8 * half;           // C/D layout row mapping
            Y[(size_t)m * BINS + n] = acc[t][r] + bv;
        }
    }
}

// ---------------------------------------------------------------------------
// Depthwise conv1d k=3 pad=1 along W, data layout [B, W, C] row-major.
// ---------------------------------------------------------------------------
__global__ void mba_dwconv3(const float* __restrict__ in, const float* __restrict__ cw,
                            const float* __restrict__ cb, float* __restrict__ out,
                            int total)
{
    const int idx = blockIdx.x * blockDim.x + threadIdx.x;
    if (idx >= total) return;
    const int c = idx & (BINS - 1);
    const int w = (idx >> 9) & (CROP - 1);
    const float x0 = (w > 0)        ? in[idx - BINS] : 0.0f;
    const float x1 = in[idx];
    const float x2 = (w < CROP - 1) ? in[idx + BINS] : 0.0f;
    out[idx] = cw[c * 3 + 0] * x0 + cw[c * 3 + 1] * x1 + cw[c * 3 + 2] * x2 + cb[c];
}

// ---------------------------------------------------------------------------
// Fused multiband attention, flash-style. One wave owns 16 query rows of one
// (b,h). Scores tile = Q.K^T + er-bias via chained WMMAs; online softmax with
// shfl_xor row reductions; P goes D-layout -> LDS -> A-layout; O += P.V.
// Writes O in [B, W, h*64+d] layout (== transpose+reshape of the reference).
// ---------------------------------------------------------------------------
__global__ __launch_bounds__(128)
void mba_attention_wmma(const float* __restrict__ Q, const float* __restrict__ Km,
                        const float* __restrict__ V, const float* __restrict__ er,
                        float* __restrict__ O)
{
    __shared__ __bf16 plds[4][16 * 32];     // per-wave P staging tile (1 KB each)

    const int lane = threadIdx.x & 31;
    const int wid  = threadIdx.x >> 5;
    const int half = lane >> 4;
    const int l16  = lane & 15;

    const int job = blockIdx.x * 4 + wid;   // 0..2047
    const int ib  = job & 63;               // query block within (b,h)
    const int bh  = job >> 6;               // 0..31
    const int b   = bh >> 3;
    const int h   = bh & 7;
    const int i0  = ib << 4;

    const size_t rbase = (size_t)b * CROP;
    const float* qbase = Q + rbase * BINS + h * HD;
    const float* kbase = Km + rbase * BINS + h * HD;
    const float* vbase = V + rbase * BINS + h * HD;

    // A fragments: Q rows i0..i0+15 (K = d, two 32-chunks) and er^T rows.
    v16bf aq0, aq1, ae0, ae1;
    {
        const float* qrow = qbase + (size_t)(i0 + l16) * BINS;
#pragma unroll
        for (int p = 0; p < 8; ++p) {
            const int k = kidx(p, half);
            const float2 f0 = *(const float2*)(qrow + k);
            const float2 f1 = *(const float2*)(qrow + 32 + k);
            aq0[2 * p] = bfc(f0.x); aq0[2 * p + 1] = bfc(f0.y);
            aq1[2 * p] = bfc(f1.x); aq1[2 * p + 1] = bfc(f1.y);
        }
        // bias[i,j] = sum_d er[d, i] * q[j, d]  ->  A[i,d] = er[d, i0+i]
#pragma unroll
        for (int p = 0; p < 8; ++p) {
            const int k = kidx(p, half);
            ae0[2 * p]     = bfc(er[(size_t)(k)      * CROP + i0 + l16]);
            ae0[2 * p + 1] = bfc(er[(size_t)(k + 1)  * CROP + i0 + l16]);
            ae1[2 * p]     = bfc(er[(size_t)(k + 32) * CROP + i0 + l16]);
            ae1[2 * p + 1] = bfc(er[(size_t)(k + 33) * CROP + i0 + l16]);
        }
    }

    float m_run[8], l_run[8];
#pragma unroll
    for (int r = 0; r < 8; ++r) { m_run[r] = -1e30f; l_run[r] = 0.0f; }
    v8f oacc[4] = {{}, {}, {}, {}};
    const float scale = 0.044194173824159216f;   // 1/sqrt(512)

    __bf16* myp = plds[wid];

    for (int j0 = 0; j0 < CROP; j0 += 32) {
        // prefetch next key-block rows (global_prefetch_b8; lane covers row j0+32+lane)
        if (j0 + 32 < CROP) {
            const size_t pr = (size_t)(j0 + 32 + lane) * BINS;
            __builtin_prefetch(kbase + pr, 0, 0);
            __builtin_prefetch(qbase + pr, 0, 0);
            __builtin_prefetch(vbase + pr, 0, 0);
        }

        v8f s[2];
#pragma unroll
        for (int t = 0; t < 2; ++t) {
            const float* krow  = kbase + (size_t)(j0 + t * 16 + l16) * BINS;
            const float* qjrow = qbase + (size_t)(j0 + t * 16 + l16) * BINS;
            v16bf bk0, bk1, bq0, bq1;
#pragma unroll
            for (int p = 0; p < 8; ++p) {
                const int k = kidx(p, half);
                const float2 fk0 = *(const float2*)(krow + k);
                const float2 fk1 = *(const float2*)(krow + 32 + k);
                const float2 fq0 = *(const float2*)(qjrow + k);
                const float2 fq1 = *(const float2*)(qjrow + 32 + k);
                bk0[2 * p] = bfc(fk0.x); bk0[2 * p + 1] = bfc(fk0.y);
                bk1[2 * p] = bfc(fk1.x); bk1[2 * p + 1] = bfc(fk1.y);
                bq0[2 * p] = bfc(fq0.x); bq0[2 * p + 1] = bfc(fq0.y);
                bq1[2 * p] = bfc(fq1.x); bq1[2 * p + 1] = bfc(fq1.y);
            }
            v8f c = {};
            c = wmma_bf16(ae0, bq0, c);   // er-bias, d = 0..31
            c = wmma_bf16(ae1, bq1, c);   // er-bias, d = 32..63
            c = wmma_bf16(aq0, bk0, c);   // Q.K^T,  d = 0..31
            c = wmma_bf16(aq1, bk1, c);   // Q.K^T,  d = 32..63
            s[t] = c;
        }

        // online softmax across the 16x32 score tile (rows map r + 8*half)
#pragma unroll
        for (int r = 0; r < 8; ++r) {
            const float s0 = s[0][r] * scale;
            const float s1 = s[1][r] * scale;
            float mx = fmaxf(s0, s1);
            mx = fmaxf(mx, __shfl_xor(mx, 1, 32));
            mx = fmaxf(mx, __shfl_xor(mx, 2, 32));
            mx = fmaxf(mx, __shfl_xor(mx, 4, 32));
            mx = fmaxf(mx, __shfl_xor(mx, 8, 32));
            const float nm = fmaxf(m_run[r], mx);
            const float alpha = __expf(m_run[r] - nm);
            const float p0 = __expf(s0 - nm);
            const float p1 = __expf(s1 - nm);
            float ls = p0 + p1;
            ls += __shfl_xor(ls, 1, 32);
            ls += __shfl_xor(ls, 2, 32);
            ls += __shfl_xor(ls, 4, 32);
            ls += __shfl_xor(ls, 8, 32);
            l_run[r] = l_run[r] * alpha + ls;
            m_run[r] = nm;
            oacc[0][r] *= alpha; oacc[1][r] *= alpha;
            oacc[2][r] *= alpha; oacc[3][r] *= alpha;
            const int m = r + 8 * half;          // D-layout row of this lane
            myp[m * 32 + l16]      = bfc(p0);
            myp[m * 32 + 16 + l16] = bfc(p1);
        }
        asm volatile("s_wait_dscnt 0" ::: "memory");

        // P back as A fragment (rows = queries, K = 32 keys)
        v16bf pa;
#pragma unroll
        for (int p = 0; p < 8; ++p) {
            const int k = kidx(p, half);
            pa[2 * p]     = myp[l16 * 32 + k];
            pa[2 * p + 1] = myp[l16 * 32 + k + 1];
        }
        asm volatile("s_wait_dscnt 0" ::: "memory");

        // O += P.V : B[k=j, n=d] = V[j0+k, dt*16+n]
#pragma unroll
        for (int dt = 0; dt < 4; ++dt) {
            const float* vcol = vbase + dt * 16 + l16;
            v16bf bv;
#pragma unroll
            for (int p = 0; p < 8; ++p) {
                const int k = kidx(p, half);
                bv[2 * p]     = bfc(vcol[(size_t)(j0 + k)     * BINS]);
                bv[2 * p + 1] = bfc(vcol[(size_t)(j0 + k + 1) * BINS]);
            }
            oacc[dt] = wmma_bf16(pa, bv, oacc[dt]);
        }
    }

    // epilogue: O[b, i, h*64 + d] = oacc / l
    float* obase = O + rbase * BINS + h * HD;
#pragma unroll
    for (int r = 0; r < 8; ++r) {
        const int m = i0 + r + 8 * half;
        const float inv = 1.0f / l_run[r];
        float* orow = obase + (size_t)m * BINS;
#pragma unroll
        for (int dt = 0; dt < 4; ++dt)
            orow[dt * 16 + l16] = oacc[dt][r] * inv;
    }
}

// ---------------------------------------------------------------------------
extern "C" void kernel_launch(void* const* d_in, const int* in_sizes, int n_in,
                              void* d_out, int out_size, void* d_ws, size_t ws_size,
                              hipStream_t stream)
{
    const float* x    = (const float*)d_in[0];
    const float* q_w  = (const float*)d_in[1];
    const float* q_b  = (const float*)d_in[2];
    const float* q_cw = (const float*)d_in[3];
    const float* q_cb = (const float*)d_in[4];
    const float* k_w  = (const float*)d_in[5];
    const float* k_b  = (const float*)d_in[6];
    const float* k_cw = (const float*)d_in[7];
    const float* k_cb = (const float*)d_in[8];
    const float* v_w  = (const float*)d_in[9];
    const float* v_b  = (const float*)d_in[10];
    const float* v_cw = (const float*)d_in[11];
    const float* v_cb = (const float*)d_in[12];
    const float* o_w  = (const float*)d_in[13];
    const float* o_b  = (const float*)d_in[14];
    const float* er   = (const float*)d_in[15];

    const size_t plane = (size_t)ROWS * BINS;     // 2M floats
    float* tmp = (float*)d_ws;
    float* qb  = tmp + plane;
    float* kb  = qb + plane;
    float* vb  = kb + plane;
    float* ao  = vb + plane;

    const int total = (int)plane;
    const int gemm_blocks = ((ROWS / 16) * (BINS / 64)) / 4;   // 512
    const int conv_blocks = (total + 255) / 256;

    mba_gemm_bias_wmma<<<gemm_blocks, 128, 0, stream>>>(x, q_w, q_b, tmp);
    mba_dwconv3<<<conv_blocks, 256, 0, stream>>>(tmp, q_cw, q_cb, qb, total);

    mba_gemm_bias_wmma<<<gemm_blocks, 128, 0, stream>>>(x, k_w, k_b, tmp);
    mba_dwconv3<<<conv_blocks, 256, 0, stream>>>(tmp, k_cw, k_cb, kb, total);

    mba_gemm_bias_wmma<<<gemm_blocks, 128, 0, stream>>>(x, v_w, v_b, tmp);
    mba_dwconv3<<<conv_blocks, 256, 0, stream>>>(tmp, v_cw, v_cb, vb, total);

    mba_attention_wmma<<<512, 128, 0, stream>>>(qb, kb, vb, er, ao);

    mba_gemm_bias_wmma<<<gemm_blocks, 128, 0, stream>>>(ao, o_w, o_b, (float*)d_out);
}
